// HyperGraphLayer_9947144258059
// MI455X (gfx1250) — matrix operand order repaired and checked by reading, not compiled, not run on hardware
//
#include <hip/hip_runtime.h>
#include <hip/hip_bf16.h>
#include <math.h>

// ---------------------------------------------------------------------------
// Problem constants (match reference)
// ---------------------------------------------------------------------------
#define NN   4096      // nodes
#define DD   256       // feature dim
#define HH   8         // heads
#define DHD  32        // head dim
#define NHE  8192      // hyperedges
#define EE   32768     // edges
#define NG   64        // graphs
#define D3   768       // 3*D (qkv)
#define D2   512       // 2*D (mlp hidden)
#define EPSN 1e-5f

#define BM 64          // GEMM block tile M (4 waves)
#define BN 32          // GEMM block tile N (2 waves)
#define BK 32          // GEMM K step (one WMMA)
#define AP 40          // LDS row pitch in halves (80B, bank-conflict-free)

typedef __attribute__((ext_vector_type(16))) __bf16 v16bf;
typedef __attribute__((ext_vector_type(8)))  __bf16 v8bf16;
typedef __attribute__((ext_vector_type(8)))  float  v8f;

// ---------------------------------------------------------------------------
// helpers
// ---------------------------------------------------------------------------
__device__ __forceinline__ unsigned short f2bf(float f) {
  unsigned int u = __float_as_uint(f);
  unsigned int r = u + 0x7FFFu + ((u >> 16) & 1u);   // round-to-nearest-even
  return (unsigned short)(r >> 16);
}
__device__ __forceinline__ __bf16 us2bf(unsigned short u) {
  __bf16 r; __builtin_memcpy(&r, &u, 2); return r;
}
__device__ __forceinline__ v16bf cat8(v8bf16 lo, v8bf16 hi) {
  return __builtin_shufflevector(lo, hi, 0,1,2,3,4,5,6,7,8,9,10,11,12,13,14,15);
}
__device__ __forceinline__ float geluf(float x) {
  return 0.5f * x * (1.0f + erff(x * 0.70710678118654752f));   // exact gelu
}
__device__ __forceinline__ float eluf(float x) { return x > 0.f ? x : expm1f(x); }

// ---------------------------------------------------------------------------
// fp32 -> bf16 convert (vectorized x4)
// ---------------------------------------------------------------------------
__global__ void f32_to_bf16_kernel(const float* __restrict__ s, unsigned short* __restrict__ d, int n4) {
  int i = blockIdx.x * blockDim.x + threadIdx.x;
  if (i >= n4) return;
  float4 v = ((const float4*)s)[i];
  ushort4 o;
  o.x = f2bf(v.x); o.y = f2bf(v.y); o.z = f2bf(v.z); o.w = f2bf(v.w);
  ((ushort4*)d)[i] = o;
}

// fp32 [K,Nc] -> bf16 transposed [Nc,K] (weights; done once, makes all GEMM
// B-fragment reads K-contiguous)
__global__ void f32_to_bf16T_kernel(const float* __restrict__ s, unsigned short* __restrict__ d, int K, int Nc) {
  int i = blockIdx.x * blockDim.x + threadIdx.x;
  if (i >= K * Nc) return;
  int k = i / Nc, n = i % Nc;
  d[(size_t)n * K + k] = f2bf(s[i]);
}

// ---------------------------------------------------------------------------
// bf16 WMMA GEMM: out = act(A[M,K] @ W[K,Nc] + bias (+ addend)), W given as
// Bt[Nc,K] (transposed).  256 threads = 8 wave32 waves, each wave one 16x16
// tile of a 64x32 block tile.  A/B tiles staged into LDS with
// global_load_async_to_lds_b128 (ASYNCcnt), fragments read as ds_load_b128.
// ---------------------------------------------------------------------------
__global__ __launch_bounds__(256) void gemm_bf16_kernel(
    const unsigned short* __restrict__ A,   // [M,K] bf16 row-major
    const unsigned short* __restrict__ Bt,  // [Nc,K] bf16 row-major (= W^T)
    const float* __restrict__ bias, const float* __restrict__ addend,
    float* __restrict__ outF, unsigned short* __restrict__ outB,
    int M, int Nc, int K, int act)
{
  __shared__ __align__(16) unsigned short As[BM][AP];
  __shared__ __align__(16) unsigned short Bs[BN][AP];
  const int tid  = threadIdx.x;
  const int lane = tid & 31;
  const int w    = tid >> 5;
  const int wrow = w & 3, wcol = w >> 2;
  const int rowB = blockIdx.y * BM;
  const int colB = blockIdx.x * BN;
  const int l16  = lane & 15;
  const int grp  = lane >> 4;

  // staging coordinates: one 16B async copy per thread (A), threads<128 (B)
  const int ar = tid >> 2;                 // 0..63
  const int ac = (tid & 3) * 8;            // halves
  const int br = (tid >> 2) & 31;          // 0..31
  const int bc = (tid & 3) * 8;
  const unsigned aoff = (unsigned)(uintptr_t)(&As[ar][ac]);
  const unsigned boff = (unsigned)(uintptr_t)(&Bs[br][bc]);
  const unsigned short* Ag = A  + (size_t)(rowB + ar) * K + ac;
  const unsigned short* Bg = Bt + (size_t)(colB + br) * K + bc;

  v8f acc = {};
  for (int k0 = 0; k0 < K; k0 += BK) {
    __syncthreads();                       // WAR: previous frag reads complete
    if (k0 + BK < K) __builtin_prefetch(Ag + k0 + BK, 0, 1);   // global_prefetch
    asm volatile("global_load_async_to_lds_b128 %0, %1, off"
                 :: "v"(aoff), "v"((unsigned long long)(uintptr_t)(Ag + k0))
                 : "memory");
    if (tid < 128)                         // wave-uniform (waves 0..3)
      asm volatile("global_load_async_to_lds_b128 %0, %1, off"
                   :: "v"(boff), "v"((unsigned long long)(uintptr_t)(Bg + k0))
                   : "memory");
    asm volatile("s_wait_asynccnt 0" ::: "memory");
    __syncthreads();                       // RAW: all waves' tiles visible

    // A frag (ISA 7.12.2): lane=M row, halves {grp*8+0..7, 16+grp*8+0..7}
    v8bf16 alo = *(const v8bf16*)(&As[wrow * 16 + l16][grp * 8]);
    v8bf16 ahi = *(const v8bf16*)(&As[wrow * 16 + l16][16 + grp * 8]);
    // B frag: lane=N col, K rows {grp*8+0..7, 16+grp*8+0..7} (same as A)
    const int kb = grp * 8;
    v8bf16 blo = *(const v8bf16*)(&Bs[wcol * 16 + l16][kb]);
    v8bf16 bhi = *(const v8bf16*)(&Bs[wcol * 16 + l16][kb + 16]);
    acc = __builtin_amdgcn_wmma_f32_16x16x32_bf16(false, cat8(alo, ahi),
                                                  false, cat8(blo, bhi),
                                                  (short)0, acc, false, false);
  }
  const int n = colB + wcol * 16 + l16;
  const float bs = bias ? bias[n] : 0.f;
#pragma unroll
  for (int v = 0; v < 8; ++v) {            // C/D frag: VGPR v -> M = v + 8*grp
    int m = rowB + wrow * 16 + v + grp * 8;
    size_t idx = (size_t)m * Nc + n;
    float val = acc[v] + bs;
    if (addend) val += addend[idx];
    if (act == 1) val = geluf(val);
    if (outF) outF[idx] = val;
    if (outB) outB[idx] = f2bf(val);
  }
}

// ---------------------------------------------------------------------------
// Flash attention over all N nodes, H=8 heads, dh=32.
// One wave per (head, 16-query tile). Keys processed 32 at a time:
//   2 WMMAs QK^T, online softmax (16-lane shfl reductions), P via LDS, 2 WMMAs P*V.
// Q/K/P fragment loads are 16B-vectorized.
// ---------------------------------------------------------------------------
__global__ __launch_bounds__(256) void flash_attn_kernel(
    const unsigned short* __restrict__ qkv,   // [N, 3D] bf16 (q|k|v)
    unsigned short* __restrict__ ctx)         // [N, D]  bf16
{
  __shared__ __align__(16) unsigned short Pl[8][16][AP];  // per-wave exp(P) tile
  const int lane = threadIdx.x & 31;
  const int w    = threadIdx.x >> 5;
  const int wid  = blockIdx.x * 8 + w;        // 0..2047
  const int head = wid >> 8;                  // 8 heads x 256 q-tiles
  const int qt   = wid & 255;
  const int l16  = lane & 15;
  const int grp  = lane >> 4;
  const float sc = 0.17677669529663687f;      // 1/sqrt(32)

  v16bf aq;                                   // Q fragment, loaded once
  {
    const unsigned short* qb = qkv + (size_t)(qt * 16 + l16) * D3 + head * DHD + grp * 8;
    aq = cat8(*(const v8bf16*)qb, *(const v8bf16*)(qb + 16));
  }
  float mi[8], li[8];
#pragma unroll
  for (int v = 0; v < 8; ++v) { mi[v] = -INFINITY; li[v] = 0.f; }
  v8f acc0 = {}, acc1 = {};

  for (int kc = 0; kc < NN; kc += 32) {
    // --- K^T fragments: lane = key col; dims {kb..kb+7, kb+16..kb+23} ---
    const int kb = grp * 8;
    const unsigned short* kp0 = qkv + (size_t)(kc + l16) * D3 + DD + head * DHD + kb;
    const unsigned short* kp1 = kp0 + (size_t)16 * D3;
    v16bf bk0 = cat8(*(const v8bf16*)kp0, *(const v8bf16*)(kp0 + 16));
    v16bf bk1 = cat8(*(const v8bf16*)kp1, *(const v8bf16*)(kp1 + 16));
    v8f z = {};
    v8f s0 = __builtin_amdgcn_wmma_f32_16x16x32_bf16(false, aq, false, bk0, (short)0, z, false, false);
    v8f s1 = __builtin_amdgcn_wmma_f32_16x16x32_bf16(false, aq, false, bk1, (short)0, z, false, false);

    // --- online softmax update; rows live across lanes 0-15 / 16-31 ---
#pragma unroll
    for (int v = 0; v < 8; ++v) {
      float x0 = s0[v] * sc, x1 = s1[v] * sc;
      float mx = fmaxf(x0, x1);
#pragma unroll
      for (int m = 1; m < 16; m <<= 1) mx = fmaxf(mx, __shfl_xor(mx, m, 32));
      float mn = fmaxf(mi[v], mx);
      float p0 = expf(x0 - mn), p1 = expf(x1 - mn);
      float rs = p0 + p1;
#pragma unroll
      for (int m = 1; m < 16; m <<= 1) rs += __shfl_xor(rs, m, 32);
      float corr = expf(mi[v] - mn);
      li[v] = li[v] * corr + rs;
      mi[v] = mn;
      acc0[v] *= corr; acc1[v] *= corr;
      int r = v + grp * 8;
      Pl[w][r][l16]      = f2bf(p0);
      Pl[w][r][16 + l16] = f2bf(p1);
    }
    asm volatile("s_wait_dscnt 0" ::: "memory");   // LDS stores visible to own wave

    // --- reload P as an A fragment (16 queries x 32 keys), vectorized ---
    const unsigned short* pp = &Pl[w][l16][grp * 16];
    v16bf pa = cat8(*(const v8bf16*)pp, *(const v8bf16*)(pp + 8));

    // --- V fragments: B = V chunk (32 keys x 16 dims), two dim halves ---
    v16bf bv0, bv1;
    {
      const int vb = grp ? 16 : 0;
#pragma unroll
      for (int v = 0; v < 8; ++v) {
        const unsigned short* vp0 = qkv + (size_t)(kc + vb + 2 * v) * D3 + 2 * DD + head * DHD;
        const unsigned short* vp1 = vp0 + D3;
        bv0[2 * v] = us2bf(vp0[l16]);      bv0[2 * v + 1] = us2bf(vp1[l16]);
        bv1[2 * v] = us2bf(vp0[16 + l16]); bv1[2 * v + 1] = us2bf(vp1[16 + l16]);
      }
    }
    acc0 = __builtin_amdgcn_wmma_f32_16x16x32_bf16(false, pa, false, bv0, (short)0, acc0, false, false);
    acc1 = __builtin_amdgcn_wmma_f32_16x16x32_bf16(false, pa, false, bv1, (short)0, acc1, false, false);
  }
#pragma unroll
  for (int v = 0; v < 8; ++v) {
    int q = qt * 16 + v + grp * 8;
    float inv = 1.0f / li[v];
    ctx[(size_t)q * DD + head * DHD + l16]      = f2bf(acc0[v] * inv);
    ctx[(size_t)q * DD + head * DHD + 16 + l16] = f2bf(acc1[v] * inv);
  }
}

// ---------------------------------------------------------------------------
// hyperedge id scan: hid = cumsum(transitions 1->0), single block Hillis-Steele
// ---------------------------------------------------------------------------
__global__ __launch_bounds__(1024) void hid_scan_kernel(const int* __restrict__ ht, int* __restrict__ hid, int n) {
  __shared__ int buf[1024];
  __shared__ int carry;
  int tid = threadIdx.x;
  if (tid == 0) carry = 0;
  __syncthreads();
  for (int base = 0; base < n; base += 1024) {
    int i = base + tid;
    int t = 0;
    if (i > 0 && i < n) t = (ht[i - 1] == 1 && ht[i] == 0) ? 1 : 0;
    buf[tid] = t;
    __syncthreads();
    for (int off = 1; off < 1024; off <<= 1) {
      int add = (tid >= off) ? buf[tid - off] : 0;
      __syncthreads();
      buf[tid] += add;
      __syncthreads();
    }
    int val = carry + buf[tid];
    if (i < n) hid[i] = val;
    __syncthreads();
    if (tid == 0) carry += buf[1023];
    __syncthreads();
  }
}

// segment boundaries of a sorted id array
__global__ void seg_bounds_kernel(const int* __restrict__ seg, int* __restrict__ st, int* __restrict__ en, int n) {
  int i = blockIdx.x * blockDim.x + threadIdx.x;
  if (i >= n) return;
  int s = seg[i];
  if (i == 0 || seg[i - 1] != s) st[s] = i;
  if (i == n - 1 || seg[i + 1] != s) en[s] = i + 1;
}

// dst[e,:] = src[idx[e],:]  (bf16 rows of width DD; 16B vectors)
__global__ void gather_rows_bf_kernel(const unsigned short* __restrict__ src, const int* __restrict__ idx,
                                      unsigned short* __restrict__ dst, int rows) {
  int i = blockIdx.x * blockDim.x + threadIdx.x;   // rows * 32 vec8-groups
  if (i >= rows * 32) return;
  int r = i >> 5, g = i & 31;
  *(uint4*)(dst + (size_t)r * DD + g * 8) = *(const uint4*)(src + (size_t)idx[r] * DD + g * 8);
}

// hetero select: out = bf16( t[row] ? a1 : a0 )
__global__ void select_bf_kernel(const float* __restrict__ a0, const float* __restrict__ a1,
                                 const int* __restrict__ t, unsigned short* __restrict__ dst, int rows) {
  int i = blockIdx.x * blockDim.x + threadIdx.x;
  if (i >= rows * DD) return;
  int r = i >> 8;
  dst[i] = f2bf(t[r] ? a1[i] : a0[i]);
}

// per-(edge,head) attention logits: dot over 32 dims
__global__ void logits_kernel(const float* __restrict__ q, const float* __restrict__ k,
                              const int* __restrict__ qidx, float* __restrict__ out, int e_count) {
  int i = blockIdx.x * blockDim.x + threadIdx.x;
  if (i >= e_count * HH) return;
  int e = i >> 3, h = i & 7;
  const float* qp = q + (size_t)qidx[e] * DD + h * DHD;
  const float* kp = k + (size_t)e * DD + h * DHD;
  float s = 0.f;
#pragma unroll
  for (int d = 0; d < DHD; ++d) s += qp[d] * kp[d];
  out[i] = s * 0.17677669529663687f;
}

// PyG-style segment softmax (segments are contiguous, from seg_bounds)
__global__ void seg_softmax_kernel(const float* __restrict__ a, const int* __restrict__ st,
                                   const int* __restrict__ en, float* __restrict__ s, int nseg) {
  int i = blockIdx.x * blockDim.x + threadIdx.x;
  if (i >= nseg * HH) return;
  int g = i >> 3, h = i & 7;
  int e0 = st[g], e1 = en[g];
  if (e0 >= e1) return;
  float m = -INFINITY;
  for (int e = e0; e < e1; ++e) m = fmaxf(m, a[e * HH + h]);
  if (!isfinite(m)) m = 0.f;
  float sum = 0.f;
  for (int e = e0; e < e1; ++e) sum += expf(a[e * HH + h] - m);
  float inv = 1.f / (sum + 1e-16f);
  for (int e = e0; e < e1; ++e) s[e * HH + h] = expf(a[e * HH + h] - m) * inv;
}

// he[g,:] += sum over segment edges of v1[e,:] * s[e,head(d)]
__global__ __launch_bounds__(256) void he_accum_kernel(const float* __restrict__ v1, const float* __restrict__ sfx,
                                                       const int* __restrict__ st, const int* __restrict__ en,
                                                       float* __restrict__ he) {
  int g = blockIdx.x, d = threadIdx.x;
  int h = d >> 5;
  float acc = 0.f;
  for (int e = st[g]; e < en[g]; ++e) acc += v1[(size_t)e * DD + d] * sfx[e * HH + h];
  he[(size_t)g * DD + d] += acc;
}

// scatter-add messages to nodes
__global__ void msg_scatter_kernel(const float* __restrict__ v2, const float* __restrict__ sfx,
                                   const int* __restrict__ nidx, float* __restrict__ agg) {
  int i = blockIdx.x * blockDim.x + threadIdx.x;
  if (i >= EE * DD) return;
  int e = i >> 8, d = i & 255;
  float val = v2[i] * sfx[e * HH + (d >> 5)];
  atomicAdd(&agg[(size_t)nidx[e] * DD + d], val);
}

// GraphNorm: per-graph mean / var / apply (+ELU)
__global__ __launch_bounds__(256) void gn_mean_kernel(const float* __restrict__ o, const int* st, const int* en,
                                                      float* __restrict__ mean) {
  int g = blockIdx.x, d = threadIdx.x;
  int a = st[g], b = en[g];
  float s = 0.f;
  for (int n = a; n < b; ++n) s += o[(size_t)n * DD + d];
  float cnt = (float)(b > a ? b - a : 1);
  mean[g * DD + d] = s / cnt;
}
__global__ __launch_bounds__(256) void gn_var_kernel(const float* __restrict__ o, const int* st, const int* en,
                                                     const float* __restrict__ mean, const float* __restrict__ gms,
                                                     float* __restrict__ var) {
  int g = blockIdx.x, d = threadIdx.x;
  int a = st[g], b = en[g];
  float mu = mean[g * DD + d] * gms[d];
  float s = 0.f;
  for (int n = a; n < b; ++n) { float c = o[(size_t)n * DD + d] - mu; s += c * c; }
  float cnt = (float)(b > a ? b - a : 1);
  var[g * DD + d] = s / cnt;
}
__global__ void gn_apply_kernel(const float* __restrict__ o, const int* __restrict__ batch,
                                const float* __restrict__ mean, const float* __restrict__ var,
                                const float* __restrict__ gw, const float* __restrict__ gb,
                                const float* __restrict__ gms, float* __restrict__ out) {
  int i = blockIdx.x * blockDim.x + threadIdx.x;
  if (i >= NN * DD) return;
  int n = i >> 8, d = i & 255;
  int g = batch[n];
  float c = o[i] - mean[g * DD + d] * gms[d];
  float val = gw[d] * c / sqrtf(var[g * DD + d] + EPSN) + gb[d];
  out[i] = eluf(val);
}

// LayerNorm over rows of width 256 (optionally adds in2 first)
__global__ __launch_bounds__(256) void ln_kernel(const float* __restrict__ in1, const float* __restrict__ in2,
                                                 const float* __restrict__ w, const float* __restrict__ b,
                                                 float* __restrict__ outF, unsigned short* __restrict__ outB) {
  int r = blockIdx.x, d = threadIdx.x;
  __shared__ float red[256];
  float v = in1[(size_t)r * DD + d];
  if (in2) v += in2[(size_t)r * DD + d];
  red[d] = v; __syncthreads();
  for (int off = 128; off > 0; off >>= 1) { if (d < off) red[d] += red[d + off]; __syncthreads(); }
  float mu = red[0] * (1.f / DD); __syncthreads();
  float c = v - mu;
  red[d] = c * c; __syncthreads();
  for (int off = 128; off > 0; off >>= 1) { if (d < off) red[d] += red[d + off]; __syncthreads(); }
  float var = red[0] * (1.f / DD);
  float o = w[d] * c / sqrtf(var + EPSN) + b[d];
  size_t idx = (size_t)r * DD + d;
  if (outF) outF[idx] = o;
  if (outB) outB[idx] = f2bf(o);
}

__global__ void add_kernel(const float* __restrict__ a, const float* __restrict__ b,
                           float* __restrict__ outF, unsigned short* __restrict__ outB, int n) {
  int i = blockIdx.x * blockDim.x + threadIdx.x;
  if (i >= n) return;
  float v = a[i] + b[i];
  outF[i] = v;
  if (outB) outB[i] = f2bf(v);
}

// ---------------------------------------------------------------------------
// host orchestration
// ---------------------------------------------------------------------------
extern "C" void kernel_launch(void* const* d_in, const int* in_sizes, int n_in,
                              void* d_out, int out_size, void* d_ws, size_t ws_size,
                              hipStream_t stream)
{
  (void)in_sizes; (void)n_in; (void)out_size; (void)ws_size;
  // ---- inputs (setup_inputs dict order) ----
  const float* x          = (const float*)d_in[0];
  const float* edge_attr  = (const float*)d_in[1];
  const float* Wq1 = (const float*)d_in[2];
  const float* Wk1 = (const float*)d_in[3];
  const float* Wv1 = (const float*)d_in[4];
  const float* We  = (const float*)d_in[5];
  const float* Wq2 = (const float*)d_in[6];
  const float* Wk2 = (const float*)d_in[7];
  const float* Wv2 = (const float*)d_in[8];
  const float* Wu1 = (const float*)d_in[9];
  const float* Wu2 = (const float*)d_in[10];
  const float* Wout = (const float*)d_in[11];
  const float* Wht  = (const float*)d_in[12];
  const float* bht  = (const float*)d_in[13];
  const float* Wtht = (const float*)d_in[14];
  const float* btht = (const float*)d_in[15];
  const float* be   = (const float*)d_in[16];
  const float* bu1  = (const float*)d_in[17];
  const float* bu2  = (const float*)d_in[18];
  const float* bout = (const float*)d_in[19];
  const float* mlp_b2 = (const float*)d_in[20];
  const float* gn_b = (const float*)d_in[21];
  const float* ln_local_b = (const float*)d_in[22];
  const float* ln_attn_b  = (const float*)d_in[23];
  const float* ln_b = (const float*)d_in[24];
  const float* gn_w = (const float*)d_in[25];
  const float* gn_ms = (const float*)d_in[26];
  const float* ln_local_w = (const float*)d_in[27];
  const float* ln_attn_w  = (const float*)d_in[28];
  const float* ln_w = (const float*)d_in[29];
  const float* attn_in_w = (const float*)d_in[30];
  const float* attn_in_b = (const float*)d_in[31];
  const float* mlp_w1 = (const float*)d_in[32];
  const float* mlp_b1 = (const float*)d_in[33];
  const float* mlp_w2 = (const float*)d_in[34];
  const int* edge_nodes = (const int*)d_in[35];
  const int* ht    = (const int*)d_in[36];
  const int* batch = (const int*)d_in[37];
  float* out = (float*)d_out;

  // ---- static workspace layout ----
  char* base = (char*)d_ws; size_t off = 0;
  auto WSA = [&](size_t bytes) -> void* {
    void* p = base + off;
    off = (off + bytes + 255) & ~(size_t)255;
    return p;
  };
  unsigned short* x_bf   = (unsigned short*)WSA((size_t)NN * DD * 2);
  unsigned short* ea_bf  = (unsigned short*)WSA((size_t)NHE * DD * 2);
  unsigned short* wq1_bf = (unsigned short*)WSA((size_t)DD * DD * 2);
  unsigned short* wk1_bf = (unsigned short*)WSA((size_t)DD * DD * 2);
  unsigned short* wv1_bf = (unsigned short*)WSA((size_t)DD * DD * 2);
  unsigned short* we_bf  = (unsigned short*)WSA((size_t)DD * DD * 2);
  unsigned short* wq2_bf = (unsigned short*)WSA((size_t)DD * DD * 2);
  unsigned short* wk2_bf = (unsigned short*)WSA((size_t)DD * DD * 2);
  unsigned short* wv2_bf = (unsigned short*)WSA((size_t)DD * DD * 2);
  unsigned short* wu1_bf = (unsigned short*)WSA((size_t)DD * DD * 2);
  unsigned short* wu2_bf = (unsigned short*)WSA((size_t)DD * DD * 2);
  unsigned short* wout_bf= (unsigned short*)WSA((size_t)DD * DD * 2);
  unsigned short* wht_bf = (unsigned short*)WSA((size_t)2 * DD * DD * 2);
  unsigned short* wtht_bf= (unsigned short*)WSA((size_t)2 * DD * DD * 2);
  unsigned short* win_bf = (unsigned short*)WSA((size_t)DD * D3 * 2);
  unsigned short* wm1_bf = (unsigned short*)WSA((size_t)DD * D2 * 2);
  unsigned short* wm2_bf = (unsigned short*)WSA((size_t)D2 * DD * 2);
  int* hid    = (int*)WSA((size_t)EE * 4);
  int* hstart = (int*)WSA((size_t)NHE * 4);
  int* hend   = (int*)WSA((size_t)NHE * 4);
  int* gstart = (int*)WSA((size_t)NG * 4);
  int* gend   = (int*)WSA((size_t)NG * 4);
  float* P0 = (float*)WSA((size_t)EE * DD * 4);     // reused: m0 / k1 / mh0 / k2
  float* P1 = (float*)WSA((size_t)EE * DD * 4);     // reused: m1 / v1 / mh1 / v2
  unsigned short* ebuf_bf = (unsigned short*)WSA((size_t)EE * DD * 2);  // xj / he-gather
  unsigned short* m_bf    = (unsigned short*)WSA((size_t)EE * DD * 2);  // m / mh
  float* q1F = (float*)WSA((size_t)NHE * DD * 4);
  float* heF = (float*)WSA((size_t)NHE * DD * 4);
  unsigned short* he_bf = (unsigned short*)WSA((size_t)NHE * DD * 2);
  float* laF = (float*)WSA((size_t)EE * HH * 4);
  float* sfx = (float*)WSA((size_t)EE * HH * 4);
  float* q2xF = (float*)WSA((size_t)NN * DD * 4);
  float* aggF = (float*)WSA((size_t)NN * DD * 4);
  unsigned short* agg_bf = (unsigned short*)WSA((size_t)NN * DD * 2);
  float* t1F  = (float*)WSA((size_t)NN * DD * 4);
  float* oF   = (float*)WSA((size_t)NN * DD * 4);
  float* meanF = (float*)WSA((size_t)NG * DD * 4);
  float* varF  = (float*)WSA((size_t)NG * DD * 4);
  float* hlpre = (float*)WSA((size_t)NN * DD * 4);
  float* hlF   = (float*)WSA((size_t)NN * DD * 4);
  unsigned short* qkv_bf = (unsigned short*)WSA((size_t)NN * D3 * 2);
  unsigned short* ctx_bf = (unsigned short*)WSA((size_t)NN * DD * 2);
  float* hatpre = (float*)WSA((size_t)NN * DD * 4);
  float* hatF   = (float*)WSA((size_t)NN * DD * 4);
  float* hsumF  = (float*)WSA((size_t)NN * DD * 4);
  unsigned short* h_bf   = (unsigned short*)WSA((size_t)NN * DD * 2);
  unsigned short* hm1_bf = (unsigned short*)WSA((size_t)NN * D2 * 2);
  float* hm2F = (float*)WSA((size_t)NN * DD * 4);

  auto conv = [&](const float* s, unsigned short* d, int n) {      // n % 4 == 0
    f32_to_bf16_kernel<<<(n / 4 + 255) / 256, 256, 0, stream>>>(s, d, n / 4);
  };
  auto convT = [&](const float* s, unsigned short* d, int K, int Ncol) {
    int n = K * Ncol;
    f32_to_bf16T_kernel<<<(n + 255) / 256, 256, 0, stream>>>(s, d, K, Ncol);
  };
  auto gemm = [&](const unsigned short* Am, const unsigned short* Btm, const float* bias, const float* addend,
                  float* oF2, unsigned short* oB, int M, int Ncol, int K, int act) {
    dim3 g(Ncol / BN, M / BM);
    gemm_bf16_kernel<<<g, 256, 0, stream>>>(Am, Btm, bias, addend, oF2, oB, M, Ncol, K, act);
  };

  // ---- precision staging: activations row-major bf16, weights transposed bf16 ----
  conv(x, x_bf, NN * DD);
  conv(edge_attr, ea_bf, NHE * DD);
  convT(Wq1, wq1_bf, DD, DD);  convT(Wk1, wk1_bf, DD, DD);  convT(Wv1, wv1_bf, DD, DD);
  convT(We, we_bf, DD, DD);    convT(Wq2, wq2_bf, DD, DD);  convT(Wk2, wk2_bf, DD, DD);
  convT(Wv2, wv2_bf, DD, DD);  convT(Wu1, wu1_bf, DD, DD);  convT(Wu2, wu2_bf, DD, DD);
  convT(Wout, wout_bf, DD, DD);
  convT(Wht, wht_bf, DD, DD);            convT(Wht + DD * DD, wht_bf + DD * DD, DD, DD);
  convT(Wtht, wtht_bf, DD, DD);          convT(Wtht + DD * DD, wtht_bf + DD * DD, DD, DD);
  convT(attn_in_w, win_bf, DD, D3);
  convT(mlp_w1, wm1_bf, DD, D2);
  convT(mlp_w2, wm2_bf, D2, DD);

  // ---- hyperedge ids + segment / graph bounds ----
  hid_scan_kernel<<<1, 1024, 0, stream>>>(ht, hid, EE);
  hipMemsetAsync(hstart, 0, (size_t)NHE * 4, stream);
  hipMemsetAsync(hend,   0, (size_t)NHE * 4, stream);
  seg_bounds_kernel<<<(EE + 255) / 256, 256, 0, stream>>>(hid, hstart, hend, EE);
  hipMemsetAsync(gstart, 0, (size_t)NG * 4, stream);
  hipMemsetAsync(gend,   0, (size_t)NG * 4, stream);
  seg_bounds_kernel<<<(NN + 255) / 256, 256, 0, stream>>>(batch, gstart, gend, NN);

  // ---- edge_update: nodes -> hyperedges ----
  gather_rows_bf_kernel<<<(EE * 32 + 255) / 256, 256, 0, stream>>>(x_bf, edge_nodes, ebuf_bf, EE);
  gemm(ebuf_bf, wht_bf,           bht,      nullptr, P0, nullptr, EE, DD, DD, 0);   // type 0
  gemm(ebuf_bf, wht_bf + DD * DD, bht + DD, nullptr, P1, nullptr, EE, DD, DD, 0);   // type 1
  select_bf_kernel<<<(EE * DD + 255) / 256, 256, 0, stream>>>(P0, P1, ht, m_bf, EE);
  gemm(m_bf, wk1_bf, nullptr, nullptr, P0, nullptr, EE, DD, DD, 0);                 // k1
  gemm(m_bf, wv1_bf, nullptr, nullptr, P1, nullptr, EE, DD, DD, 0);                 // v1
  gemm(ea_bf, wq1_bf, nullptr, nullptr, q1F, nullptr, NHE, DD, DD, 0);              // q1
  logits_kernel<<<(EE * HH + 255) / 256, 256, 0, stream>>>(q1F, P0, hid, laF, EE);
  seg_softmax_kernel<<<(NHE * HH + 255) / 256, 256, 0, stream>>>(laF, hstart, hend, sfx, NHE);
  gemm(ea_bf, we_bf, be, nullptr, heF, nullptr, NHE, DD, DD, 0);                    // edge_attr@We + be
  he_accum_kernel<<<NHE, 256, 0, stream>>>(P1, sfx, hstart, hend, heF);
  conv(heF, he_bf, NHE * DD);

  // ---- message: hyperedges -> nodes ----
  gather_rows_bf_kernel<<<(EE * 32 + 255) / 256, 256, 0, stream>>>(he_bf, hid, ebuf_bf, EE);
  gemm(ebuf_bf, wtht_bf,           btht,      nullptr, P0, nullptr, EE, DD, DD, 0);
  gemm(ebuf_bf, wtht_bf + DD * DD, btht + DD, nullptr, P1, nullptr, EE, DD, DD, 0);
  select_bf_kernel<<<(EE * DD + 255) / 256, 256, 0, stream>>>(P0, P1, ht, m_bf, EE);
  gemm(m_bf, wk2_bf, nullptr, nullptr, P0, nullptr, EE, DD, DD, 0);                 // k2
  gemm(m_bf, wv2_bf, nullptr, nullptr, P1, nullptr, EE, DD, DD, 0);                 // v2
  gemm(x_bf, wq2_bf, nullptr, nullptr, q2xF, nullptr, NN, DD, DD, 0);               // x@Wq2
  logits_kernel<<<(EE * HH + 255) / 256, 256, 0, stream>>>(q2xF, P0, edge_nodes, laF, EE);
  seg_softmax_kernel<<<(NHE * HH + 255) / 256, 256, 0, stream>>>(laF, hstart, hend, sfx, NHE);
  hipMemsetAsync(aggF, 0, (size_t)NN * DD * 4, stream);
  msg_scatter_kernel<<<(EE * DD + 255) / 256, 256, 0, stream>>>(P1, sfx, edge_nodes, aggF);

  // ---- update: u2(agg)+u1(x), GraphNorm, ELU, LN(.+x) ----
  conv(aggF, agg_bf, NN * DD);
  gemm(x_bf, wu1_bf, bu1, nullptr, t1F, nullptr, NN, DD, DD, 0);
  gemm(agg_bf, wu2_bf, bu2, t1F, oF, nullptr, NN, DD, DD, 0);
  gn_mean_kernel<<<NG, 256, 0, stream>>>(oF, gstart, gend, meanF);
  gn_var_kernel<<<NG, 256, 0, stream>>>(oF, gstart, gend, meanF, gn_ms, varF);
  gn_apply_kernel<<<(NN * DD + 255) / 256, 256, 0, stream>>>(oF, batch, meanF, varF, gn_w, gn_b, gn_ms, hlpre);
  ln_kernel<<<NN, 256, 0, stream>>>(hlpre, x, ln_local_w, ln_local_b, hlF, nullptr);

  // ---- global self-attention (flash) + out-proj + LN(.+x) ----
  gemm(x_bf, win_bf, attn_in_b, nullptr, nullptr, qkv_bf, NN, D3, DD, 0);
  flash_attn_kernel<<<256, 256, 0, stream>>>(qkv_bf, ctx_bf);
  gemm(ctx_bf, wout_bf, bout, x, hatpre, nullptr, NN, DD, DD, 0);
  ln_kernel<<<NN, 256, 0, stream>>>(hatpre, nullptr, ln_attn_w, ln_attn_b, hatF, nullptr);

  // ---- combine + MLP + final LN ----
  add_kernel<<<(NN * DD + 255) / 256, 256, 0, stream>>>(hlF, hatF, hsumF, h_bf, NN * DD);
  gemm(h_bf, wm1_bf, mlp_b1, nullptr, nullptr, hm1_bf, NN, D2, DD, 1);   // gelu
  gemm(hm1_bf, wm2_bf, mlp_b2, nullptr, hm2F, nullptr, NN, DD, D2, 1);   // gelu
  ln_kernel<<<NN, 256, 0, stream>>>(hm2F, hsumF, ln_w, ln_b, out, nullptr);
}
